// SS_HF_RNNAgent_5394478924308
// MI455X (gfx1250) — matrix operand order repaired and checked by reading, not compiled, not run on hardware
//
#include <hip/hip_runtime.h>

// ---------------------------------------------------------------------------
// SS_HF_RNNAgent forward for MI455X (gfx1250, wave32, WMMA).
// All H=256-wide dense layers run as f16-in / f32-accum WMMA GEMMs
// (v_wmma_f32_16x16x32_f16); each wave computes a 16x64 output tile (4 WMMA
// accumulators sharing one A fragment per K-step) to cut activation traffic
// 4x and raise arithmetic intensity. Small/irregular stages run as wave32
// VALU kernels with shuffle reductions.
// ---------------------------------------------------------------------------

#define BN 8192
#define LA 15
#define LE 24
#define TA (BN * LA)   // 122880 ally tokens
#define TE (BN * LE)   // 196608 enemy tokens
#define HDIM 256

typedef __attribute__((ext_vector_type(16))) _Float16 v16h;
typedef __attribute__((ext_vector_type(8)))  _Float16 v8h;
typedef __attribute__((ext_vector_type(8)))  float    v8f;

__device__ __forceinline__ float sigm(float x) { return 1.f / (1.f + __expf(-x)); }

// ---------------------------------------------------------------------------
// Generic WMMA GEMM: C[M,N] = epi(A[M,256] @ W[256,N] + bias[N])
// A: f16 row-major [M,256]. Wt: f16 row-major TRANSPOSED weights [N,256].
// Writes fp32 (Cf) and/or f16 (Ch). Requires M%32==0, N%256==0.
// Block: 256 threads = 8 waves; wave = (mstripe 0..1, ngroup 0..3);
// each wave owns a 16x64 output tile (4 x 16x16 WMMA accumulators).
// epi: 0 = none, 1 = FiLM gamma (1 + 0.5*tanh(x)).
// ---------------------------------------------------------------------------
__global__ void gemm_wmma_kernel(const _Float16* __restrict__ A,
                                 const _Float16* __restrict__ Wt,
                                 const float* __restrict__ bias,
                                 float* __restrict__ Cf,
                                 _Float16* __restrict__ Ch,
                                 int M, int N, int epi) {
  const int K = 256;
  const int wave = threadIdx.x >> 5;
  const int lane = threadIdx.x & 31;
  const int hi   = lane >> 4;
  const int l15  = lane & 15;
  const int m0 = blockIdx.x * 32 + (wave >> 2) * 16;
  const int n0 = blockIdx.y * 256 + (wave & 3) * 64;

  const _Float16* arow = A  + (size_t)(m0 + l15) * K;
  const _Float16* brow = Wt + (size_t)(n0 + l15) * K;

  v8f acc[4] = {{}, {}, {}, {}};
  #pragma unroll
  for (int k0 = 0; k0 < K; k0 += 32) {
    // A fragment (16x32, ISA 7.12.2): halves 0..7 -> K = k0+hi*8+j,
    //                                 halves 8..15 -> K = k0+16+hi*8+j
    v8h a0 = *(const v8h*)(arow + k0 + hi * 8);
    v8h a1 = *(const v8h*)(arow + k0 + 16 + hi * 8);
    v16h af;
    #pragma unroll
    for (int j = 0; j < 8; ++j) { af[j] = a0[j]; af[j + 8] = a1[j]; }
    // Four B fragments (32x16 each): same K order as A (k0+hi*8+j, k0+16+hi*8+j)
    #pragma unroll
    for (int t = 0; t < 4; ++t) {
      const _Float16* bp = brow + (size_t)(t * 16) * K + k0 + hi * 8;
      v8h b0 = *(const v8h*)(bp);
      v8h b1 = *(const v8h*)(bp + 16);
      v16h bf;
      #pragma unroll
      for (int j = 0; j < 8; ++j) { bf[j] = b0[j]; bf[j + 8] = b1[j]; }
      acc[t] = __builtin_amdgcn_wmma_f32_16x16x32_f16(
          false, af, false, bf, (short)0, acc[t], false, false);
    }
  }
  // C/D layout: acc[t][r] -> M = m0 + r + 8*hi, N = n0 + t*16 + l15
  #pragma unroll
  for (int t = 0; t < 4; ++t) {
    const int n = n0 + t * 16 + l15;
    const float bv = bias ? bias[n] : 0.f;
    #pragma unroll
    for (int r = 0; r < 8; ++r) {
      int m = m0 + r + 8 * hi;
      float v = acc[t][r] + bv;
      if (epi == 1) v = 1.f + 0.5f * tanhf(v);
      if (Cf) Cf[(size_t)m * N + n] = v;
      if (Ch) Ch[(size_t)m * N + n] = (_Float16)v;
    }
  }
}

// ---------------------------------------------------------------------------
// Weight convert + transpose: Wt[n*K+k] = (f16)W[k*N+n]
// ---------------------------------------------------------------------------
__global__ void convert_wt_kernel(const float* __restrict__ W,
                                  _Float16* __restrict__ Wt, int K, int N) {
  int idx = blockIdx.x * 256 + threadIdx.x;
  if (idx >= K * N) return;
  int k = idx / N, n = idx - k * N;
  Wt[(size_t)n * K + k] = (_Float16)W[idx];
}

__global__ void f32_to_f16_kernel(const float* __restrict__ X,
                                  _Float16* __restrict__ Y, int n) {
  int i = blockIdx.x * 256 + threadIdx.x;
  if (i < n) Y[i] = (_Float16)X[i];
}

// ---------------------------------------------------------------------------
// Prep: own=concat(own_feats, idx, act); ctx = own@Wc+bc; e_own = own@Wo+bo;
// LoRA scale vectors s = ctx@sW+sb for adapt_q / adapt_kE / adapt_kA.
// One 64-thread block per batch element.
// ---------------------------------------------------------------------------
__global__ void prep_kernel(const float* __restrict__ own_feats,
                            const int* __restrict__ agent_idx,
                            const int* __restrict__ last_act,
                            const float* __restrict__ Wc, const float* __restrict__ bc,
                            const float* __restrict__ Wo, const float* __restrict__ bo,
                            const float* __restrict__ Wsq, const float* __restrict__ bsq,
                            const float* __restrict__ WsE, const float* __restrict__ bsE,
                            const float* __restrict__ WsA, const float* __restrict__ bsA,
                            float* __restrict__ ctx, _Float16* __restrict__ ctx_h,
                            float* __restrict__ eown, float* __restrict__ svec) {
  int e = blockIdx.x;
  int t = threadIdx.x;  // 0..63
  __shared__ float own[12];
  __shared__ float sctx[256];
  if (t < 10) own[t] = own_feats[e * 10 + t];
  if (t == 10) own[10] = (float)agent_idx[e];
  if (t == 11) own[11] = (float)last_act[e];
  __syncthreads();
  #pragma unroll
  for (int i = 0; i < 4; ++i) {
    int c = t + 64 * i;
    float ac = bc[c], ao = bo[c];
    #pragma unroll
    for (int k = 0; k < 12; ++k) {
      ac += own[k] * Wc[k * 256 + c];
      ao += own[k] * Wo[k * 256 + c];
    }
    sctx[c] = ac;
    ctx [(size_t)e * 256 + c] = ac;
    ctx_h[(size_t)e * 256 + c] = (_Float16)ac;
    eown[(size_t)e * 256 + c] = ao;
  }
  __syncthreads();
  if (t < 12) {  // svec layout per element: [0..3]=q, [4..7]=kE, [8..11]=kA
    int which = t >> 2, r = t & 3;
    const float* W = which == 0 ? Wsq : (which == 1 ? WsE : WsA);
    const float* b = which == 0 ? bsq : (which == 1 ? bsE : bsA);
    float a = b[r];
    for (int k = 0; k < 256; ++k) a += sctx[k] * W[k * 4 + r];
    svec[(size_t)e * 12 + t] = a;
  }
}

// ---------------------------------------------------------------------------
// Token embed (9->256) + FiLM + rank-4 LoRA -> f16, plus valid mask.
// One wave per token; lane owns 8 columns (c = lane + 32*i).
// ---------------------------------------------------------------------------
__global__ void embed_film_lora_kernel(const float* __restrict__ F9,   // [BN*L,9]
                                       const float* __restrict__ W,    // [9,256]
                                       const float* __restrict__ b,    // [256]
                                       const float* __restrict__ gamma,// [BN,256]
                                       const float* __restrict__ U,    // [256,4]
                                       const float* __restrict__ V,    // [256,4]
                                       const float* __restrict__ svec, int sbase,
                                       _Float16* __restrict__ Out,     // [BN*L,256]
                                       float* __restrict__ maskf,      // [BN*L]
                                       int L) {
  int tok  = blockIdx.x * 8 + (threadIdx.x >> 5);
  int lane = threadIdx.x & 31;
  int e    = tok / L;
  float f[9];
  bool nz = false;
  #pragma unroll
  for (int d = 0; d < 9; ++d) { f[d] = F9[(size_t)tok * 9 + d]; nz |= (f[d] != 0.f); }
  float x[8];
  float cr[4] = {0.f, 0.f, 0.f, 0.f};
  #pragma unroll
  for (int i = 0; i < 8; ++i) {
    int c = lane + 32 * i;
    float a = b[c];
    #pragma unroll
    for (int d = 0; d < 9; ++d) a += f[d] * W[d * 256 + c];
    a *= gamma[(size_t)e * 256 + c];
    x[i] = a;
    #pragma unroll
    for (int r = 0; r < 4; ++r) cr[r] += a * V[c * 4 + r];
  }
  #pragma unroll
  for (int off = 16; off; off >>= 1)
    #pragma unroll
    for (int r = 0; r < 4; ++r) cr[r] += __shfl_xor(cr[r], off, 32);
  float cs[4];
  #pragma unroll
  for (int r = 0; r < 4; ++r) cs[r] = cr[r] * svec[(size_t)e * 12 + sbase + r];
  #pragma unroll
  for (int i = 0; i < 8; ++i) {
    int c = lane + 32 * i;
    float d = 0.f;
    #pragma unroll
    for (int r = 0; r < 4; ++r) d += U[c * 4 + r] * cs[r];
    Out[(size_t)tok * 256 + c] = (_Float16)(x[i] + d);
  }
  if (lane == 0) maskf[tok] = nz ? 1.f : 0.f;
}

// ---------------------------------------------------------------------------
// Rank-4 LoRA over fp32 rows -> f16 (used for e_own_q and hq_ptr).
// One wave per row; row index == batch element.
// ---------------------------------------------------------------------------
__global__ void lora_rows_kernel(const float* __restrict__ X,  // [BN,256]
                                 const float* __restrict__ U,
                                 const float* __restrict__ V,
                                 const float* __restrict__ svec, int sbase,
                                 _Float16* __restrict__ Out) {
  int row  = blockIdx.x * 8 + (threadIdx.x >> 5);
  int lane = threadIdx.x & 31;
  float x[8];
  float cr[4] = {0.f, 0.f, 0.f, 0.f};
  #pragma unroll
  for (int i = 0; i < 8; ++i) {
    int c = lane + 32 * i;
    x[i] = X[(size_t)row * 256 + c];
    #pragma unroll
    for (int r = 0; r < 4; ++r) cr[r] += x[i] * V[c * 4 + r];
  }
  #pragma unroll
  for (int off = 16; off; off >>= 1)
    #pragma unroll
    for (int r = 0; r < 4; ++r) cr[r] += __shfl_xor(cr[r], off, 32);
  float cs[4];
  #pragma unroll
  for (int r = 0; r < 4; ++r) cs[r] = cr[r] * svec[(size_t)row * 12 + sbase + r];
  #pragma unroll
  for (int i = 0; i < 8; ++i) {
    int c = lane + 32 * i;
    float d = 0.f;
    #pragma unroll
    for (int r = 0; r < 4; ++r) d += U[c * 4 + r] * cs[r];
    Out[(size_t)row * 256 + c] = (_Float16)(x[i] + d);
  }
}

// ---------------------------------------------------------------------------
// Single-query MHA core (L<=24, 4 heads of 64). One wave per (element, head).
// ---------------------------------------------------------------------------
__global__ void attn_kernel(const float* __restrict__ Q,     // [BN,256]
                            const _Float16* __restrict__ Kh, // [BN*L,256]
                            const _Float16* __restrict__ Vh, // [BN*L,256]
                            const float* __restrict__ maskf, // [BN*L]
                            _Float16* __restrict__ Oh,       // [BN,256]
                            int L) {
  int gid  = blockIdx.x * 8 + (threadIdx.x >> 5);
  int lane = threadIdx.x & 31;
  int e = gid >> 2;
  int h = gid & 3;
  const float* qrow = Q + (size_t)e * 256 + h * 64;
  float s;
  if (lane < L) {
    const _Float16* krow = Kh + ((size_t)e * L + lane) * 256 + h * 64;
    float acc = 0.f;
    #pragma unroll
    for (int d = 0; d < 64; ++d) acc += qrow[d] * (float)krow[d];
    s = acc * 0.125f;                                // / sqrt(64)
    if (maskf[(size_t)e * L + lane] == 0.f) s = -1e9f;
  } else {
    s = -INFINITY;                                   // excluded from softmax
  }
  float mx = s;
  #pragma unroll
  for (int off = 16; off; off >>= 1) mx = fmaxf(mx, __shfl_xor(mx, off, 32));
  float a = __expf(s - mx);
  float sum = a;
  #pragma unroll
  for (int off = 16; off; off >>= 1) sum += __shfl_xor(sum, off, 32);
  a /= sum;
  int d0 = lane * 2;
  float o0 = 0.f, o1 = 0.f;
  for (int j = 0; j < L; ++j) {
    float aj = __shfl(a, j, 32);
    const _Float16* vrow = Vh + ((size_t)e * L + j) * 256 + h * 64 + d0;
    o0 += aj * (float)vrow[0];
    o1 += aj * (float)vrow[1];
  }
  Oh[(size_t)e * 256 + h * 64 + d0]     = (_Float16)o0;
  Oh[(size_t)e * 256 + h * 64 + d0 + 1] = (_Float16)o1;
}

// ---------------------------------------------------------------------------
// Gate + fuse: g = sigmoid([zA,zE,e_own] @ Wg + bg); z = g*zE + (1-g)*zA -> f16
// ---------------------------------------------------------------------------
__global__ void gate_fuse_kernel(const float* __restrict__ zA,
                                 const float* __restrict__ zE,
                                 const float* __restrict__ eown,
                                 const float* __restrict__ Wg,  // [768]
                                 const float* __restrict__ bg,  // [1]
                                 _Float16* __restrict__ z_h) {
  int e    = blockIdx.x * 8 + (threadIdx.x >> 5);
  int lane = threadIdx.x & 31;
  float p = 0.f;
  for (int k = lane; k < 256; k += 32) {
    p += zA[(size_t)e * 256 + k] * Wg[k]
       + zE[(size_t)e * 256 + k] * Wg[256 + k]
       + eown[(size_t)e * 256 + k] * Wg[512 + k];
  }
  #pragma unroll
  for (int off = 16; off; off >>= 1) p += __shfl_xor(p, off, 32);
  float g = sigm(p + bg[0]);
  #pragma unroll
  for (int i = 0; i < 8; ++i) {
    int c = lane + 32 * i;
    float za = zA[(size_t)e * 256 + c], ze = zE[(size_t)e * 256 + c];
    z_h[(size_t)e * 256 + c] = (_Float16)(g * ze + (1.f - g) * za);
  }
}

// ---------------------------------------------------------------------------
// GRU pointwise + q_normal (256->6) + attack_bias (256->1).
// ---------------------------------------------------------------------------
__global__ void gru_kernel(const float* __restrict__ gi,   // [BN,768]
                           const float* __restrict__ gh,   // [BN,768]
                           const float* __restrict__ hprev,// [BN,256]
                           const float* __restrict__ Wn, const float* __restrict__ bn,
                           const float* __restrict__ Wab, const float* __restrict__ bab,
                           float* __restrict__ h_out,      // [BN,256] (d_out)
                           float* __restrict__ q_out,      // [BN,30]  (d_out)
                           float* __restrict__ ab_out) {   // [BN]
  int e    = blockIdx.x * 8 + (threadIdx.x >> 5);
  int lane = threadIdx.x & 31;
  float pn[6] = {0.f, 0.f, 0.f, 0.f, 0.f, 0.f};
  float pab = 0.f;
  #pragma unroll
  for (int i = 0; i < 8; ++i) {
    int c = lane + 32 * i;
    size_t b7 = (size_t)e * 768 + c;
    float r  = sigm(gi[b7]       + gh[b7]);
    float zg = sigm(gi[b7 + 256] + gh[b7 + 256]);
    float n  = tanhf(gi[b7 + 512] + r * gh[b7 + 512]);
    float hp = hprev[(size_t)e * 256 + c];
    float hn = (1.f - zg) * n + zg * hp;
    h_out[(size_t)e * 256 + c] = hn;
    #pragma unroll
    for (int o = 0; o < 6; ++o) pn[o] += hn * Wn[c * 6 + o];
    pab += hn * Wab[c];
  }
  #pragma unroll
  for (int off = 16; off; off >>= 1) {
    #pragma unroll
    for (int o = 0; o < 6; ++o) pn[o] += __shfl_xor(pn[o], off, 32);
    pab += __shfl_xor(pab, off, 32);
  }
  if (lane == 0) {
    #pragma unroll
    for (int o = 0; o < 6; ++o) q_out[(size_t)e * 30 + o] = pn[o] + bn[o];
    ab_out[e] = pab + bab[0];
  }
}

// ---------------------------------------------------------------------------
// Pointer scores (mean over 4 heads == dot/32), top-8 keep, q_attack write.
// One wave per element; lane j<24 owns enemy j.
// ---------------------------------------------------------------------------
__global__ void pointer_final_kernel(const float* __restrict__ qP,     // [BN,256]
                                     const _Float16* __restrict__ kP,  // [BN*24,256]
                                     const float* __restrict__ ef,     // [BN*24,9]
                                     const float* __restrict__ maskEf, // [BN*24]
                                     const float* __restrict__ ab,     // [BN]
                                     float* __restrict__ q_out) {      // [BN,30]
  __shared__ float sc[8][24];
  __shared__ unsigned keepm[8];
  int w    = threadIdx.x >> 5;
  int lane = threadIdx.x & 31;
  int e    = blockIdx.x * 8 + w;
  float pv = 0.f;
  bool  m  = false;
  if (lane < 24) {
    size_t t = (size_t)e * 24 + lane;
    const _Float16* krow = kP + t * 256;
    const float*    qrow = qP + (size_t)e * 256;
    float acc = 0.f;
    for (int d = 0; d < 256; ++d) acc += qrow[d] * (float)krow[d];
    pv = acc * (1.f / 32.f);   // /sqrt(64) then mean over 4 heads
    m  = (maskEf[t] != 0.f);
    float avail = ef[t * 9 + 0], dist = ef[t * 9 + 1];
    float hp    = ef[t * 9 + 4], sh   = ef[t * 9 + 5];
    float score = 2.f * avail - dist + 0.5f * (1.f - hp) + 0.25f * (1.f - sh);
    sc[w][lane] = m ? score : -1e9f;
  }
  __syncthreads();
  if (lane == 0) {  // serial top-8, ties -> lowest index (matches lax.top_k)
    unsigned taken = 0;
    for (int t = 0; t < 8; ++t) {
      float best = -INFINITY; int bi = 0;
      for (int j = 0; j < 24; ++j)
        if (!((taken >> j) & 1u) && sc[w][j] > best) { best = sc[w][j]; bi = j; }
      taken |= 1u << bi;
    }
    keepm[w] = taken;
  }
  __syncthreads();
  if (lane < 24) {
    bool keep = ((keepm[w] >> lane) & 1u) && m;
    q_out[(size_t)e * 30 + 6 + lane] = keep ? (pv + ab[e]) : -1e9f;
  }
}

// ---------------------------------------------------------------------------
// Host launch
// ---------------------------------------------------------------------------
extern "C" void kernel_launch(void* const* d_in, const int* in_sizes, int n_in,
                              void* d_out, int out_size, void* d_ws, size_t ws_size,
                              hipStream_t stream) {
  if (n_in < 61) return;
  // Input map (recursive insertion-order flatten of setup_inputs()):
  const float* own_feats   = (const float*)d_in[1];
  const float* ally_feats  = (const float*)d_in[2];
  const float* enemy_feats = (const float*)d_in[3];
  const int*   agent_idx   = (const int*)d_in[4];
  const int*   last_act    = (const int*)d_in[5];
  const float* hidden      = (const float*)d_in[6];
  auto F = [&](int i) { return (const float*)d_in[i]; };
  // 7/8 embed_own.W/b   9/10 embed_enemy  11/12 embed_ally  13/14 context
  // 15/16 film_enems    17/18 film_allies
  // 19..22 adapt_q{U,V,sW,sb}  23..26 adapt_kE  27..30 adapt_kA
  // 31..38 attn_enems{q,k,v,o}{W,b}   39..46 attn_allies{...}
  // 47/48 gate  49..52 gru{w_ih,b_ih,w_hh,b_hh}
  // 53/54 normal  55/56 attack_bias  57/58 pointer.q  59/60 pointer.k

  float* out_q = (float*)d_out;               // [BN,30]
  float* out_h = out_q + (size_t)BN * 30;     // [BN,256] == next_hidden

  // ---- workspace carve ----
  char* p = (char*)d_ws;
  auto alloc = [&](size_t bytes) -> void* {
    void* r = (void*)p; p += (bytes + 255) & ~(size_t)255; return r;
  };
  float*     ctx_f   = (float*)alloc((size_t)BN * 256 * 4);
  _Float16*  ctx_h   = (_Float16*)alloc((size_t)BN * 256 * 2);
  float*     eown_f  = (float*)alloc((size_t)BN * 256 * 4);
  float*     svec    = (float*)alloc((size_t)BN * 12 * 4);
  float*     gammaA  = (float*)alloc((size_t)BN * 256 * 4);
  float*     gammaE  = (float*)alloc((size_t)BN * 256 * 4);
  float*     maskAf  = (float*)alloc((size_t)TA * 4);
  float*     maskEf  = (float*)alloc((size_t)TE * 4);
  _Float16*  xA_h    = (_Float16*)alloc((size_t)TA * 256 * 2);
  _Float16*  xE_h    = (_Float16*)alloc((size_t)TE * 256 * 2);
  _Float16*  eownq_h = (_Float16*)alloc((size_t)BN * 256 * 2);
  // transposed f16 weights
  _Float16* wtFilmE = (_Float16*)alloc(256 * 256 * 2);
  _Float16* wtFilmA = (_Float16*)alloc(256 * 256 * 2);
  _Float16* wtQE = (_Float16*)alloc(256 * 256 * 2);
  _Float16* wtKE = (_Float16*)alloc(256 * 256 * 2);
  _Float16* wtVE = (_Float16*)alloc(256 * 256 * 2);
  _Float16* wtOE = (_Float16*)alloc(256 * 256 * 2);
  _Float16* wtQA = (_Float16*)alloc(256 * 256 * 2);
  _Float16* wtKA = (_Float16*)alloc(256 * 256 * 2);
  _Float16* wtVA = (_Float16*)alloc(256 * 256 * 2);
  _Float16* wtOA = (_Float16*)alloc(256 * 256 * 2);
  _Float16* wtPQ = (_Float16*)alloc(256 * 256 * 2);
  _Float16* wtPK = (_Float16*)alloc(256 * 256 * 2);
  _Float16* wtWIH = (_Float16*)alloc(256 * 768 * 2);
  _Float16* wtWHH = (_Float16*)alloc(256 * 768 * 2);
  float*    qA   = (float*)alloc((size_t)BN * 256 * 4);
  float*    qE   = (float*)alloc((size_t)BN * 256 * 4);
  _Float16* kA_h = (_Float16*)alloc((size_t)TA * 256 * 2);
  _Float16* vA_h = (_Float16*)alloc((size_t)TA * 256 * 2);
  _Float16* kE_h = (_Float16*)alloc((size_t)TE * 256 * 2);
  _Float16* vE_h = (_Float16*)alloc((size_t)TE * 256 * 2);
  _Float16* kP_h = (_Float16*)alloc((size_t)TE * 256 * 2);
  _Float16* oA_h = (_Float16*)alloc((size_t)BN * 256 * 2);
  _Float16* oE_h = (_Float16*)alloc((size_t)BN * 256 * 2);
  float*    zA   = (float*)alloc((size_t)BN * 256 * 4);
  float*    zE   = (float*)alloc((size_t)BN * 256 * 4);
  _Float16* z_h  = (_Float16*)alloc((size_t)BN * 256 * 2);
  _Float16* h_h  = (_Float16*)alloc((size_t)BN * 256 * 2);
  float*    gi   = (float*)alloc((size_t)BN * 768 * 4);
  float*    gh   = (float*)alloc((size_t)BN * 768 * 4);
  _Float16* hqp_h = (_Float16*)alloc((size_t)BN * 256 * 2);
  float*    qPf  = (float*)alloc((size_t)BN * 256 * 4);
  float*    ab   = (float*)alloc((size_t)BN * 4);
  (void)ws_size; (void)in_sizes; (void)out_size;

  auto conv = [&](const float* W, _Float16* Wt, int K, int N) {
    int total = K * N;
    convert_wt_kernel<<<(total + 255) / 256, 256, 0, stream>>>(W, Wt, K, N);
  };
  auto gemm = [&](const _Float16* A, const _Float16* Wt, const float* bias,
                  float* Cf, _Float16* Ch, int M, int N, int epi) {
    dim3 g(M / 32, N / 256);
    gemm_wmma_kernel<<<g, 256, 0, stream>>>(A, Wt, bias, Cf, Ch, M, N, epi);
  };

  // ---- weight conversion + hidden f16 ----
  conv(F(15), wtFilmE, 256, 256); conv(F(17), wtFilmA, 256, 256);
  conv(F(31), wtQE, 256, 256);    conv(F(33), wtKE, 256, 256);
  conv(F(35), wtVE, 256, 256);    conv(F(37), wtOE, 256, 256);
  conv(F(39), wtQA, 256, 256);    conv(F(41), wtKA, 256, 256);
  conv(F(43), wtVA, 256, 256);    conv(F(45), wtOA, 256, 256);
  conv(F(57), wtPQ, 256, 256);    conv(F(59), wtPK, 256, 256);
  conv(F(49), wtWIH, 256, 768);   conv(F(51), wtWHH, 256, 768);
  f32_to_f16_kernel<<<(BN * 256) / 256, 256, 0, stream>>>(hidden, h_h, BN * 256);

  // ---- prep: own/ctx/e_own/svec ----
  prep_kernel<<<BN, 64, 0, stream>>>(own_feats, agent_idx, last_act,
      F(13), F(14), F(7), F(8),
      F(21), F(22), F(25), F(26), F(29), F(30),
      ctx_f, ctx_h, eown_f, svec);

  // ---- FiLM gammas (WMMA, epi = 1 + 0.5*tanh) ----
  gemm(ctx_h, wtFilmA, F(18), gammaA, nullptr, BN, 256, 1);
  gemm(ctx_h, wtFilmE, F(16), gammaE, nullptr, BN, 256, 1);

  // ---- token embed + FiLM + LoRA (-> f16) ----
  embed_film_lora_kernel<<<TA / 8, 256, 0, stream>>>(
      ally_feats, F(11), F(12), gammaA, F(27), F(28), svec, 8, xA_h, maskAf, LA);
  embed_film_lora_kernel<<<TE / 8, 256, 0, stream>>>(
      enemy_feats, F(9), F(10), gammaE, F(23), F(24), svec, 4, xE_h, maskEf, LE);

  // ---- e_own_q = lora(e_own, adapt_q) ----
  lora_rows_kernel<<<BN / 8, 256, 0, stream>>>(eown_f, F(19), F(20), svec, 0, eownq_h);

  // ---- projections (WMMA) ----
  gemm(eownq_h, wtQA, F(40), qA, nullptr, BN, 256, 0);
  gemm(eownq_h, wtQE, F(32), qE, nullptr, BN, 256, 0);
  gemm(xA_h, wtKA, F(42), nullptr, kA_h, TA, 256, 0);
  gemm(xA_h, wtVA, F(44), nullptr, vA_h, TA, 256, 0);
  gemm(xE_h, wtKE, F(34), nullptr, kE_h, TE, 256, 0);
  gemm(xE_h, wtVE, F(36), nullptr, vE_h, TE, 256, 0);
  gemm(xE_h, wtPK, F(60), nullptr, kP_h, TE, 256, 0);   // pointer k (== eE_ptr_k path)

  // ---- attention (softmax core) + output projection ----
  attn_kernel<<<(BN * 4) / 8, 256, 0, stream>>>(qA, kA_h, vA_h, maskAf, oA_h, LA);
  attn_kernel<<<(BN * 4) / 8, 256, 0, stream>>>(qE, kE_h, vE_h, maskEf, oE_h, LE);
  gemm(oA_h, wtOA, F(46), zA, nullptr, BN, 256, 0);
  gemm(oE_h, wtOE, F(38), zE, nullptr, BN, 256, 0);

  // ---- gate + fuse -> z ----
  gate_fuse_kernel<<<BN / 8, 256, 0, stream>>>(zA, zE, eown_f, F(47), F(48), z_h);

  // ---- GRU: gi/gh (WMMA, N=768) + pointwise ----
  gemm(z_h, wtWIH, F(50), gi, nullptr, BN, 768, 0);
  gemm(h_h, wtWHH, F(52), gh, nullptr, BN, 768, 0);
  gru_kernel<<<BN / 8, 256, 0, stream>>>(gi, gh, hidden, F(53), F(54), F(55), F(56),
                                         out_h, out_q, ab);

  // ---- pointer: hq_ptr = lora(h_next, adapt_q); qP = hq_ptr @ Wq_ptr ----
  lora_rows_kernel<<<BN / 8, 256, 0, stream>>>(out_h, F(19), F(20), svec, 0, hqp_h);
  gemm(hqp_h, wtPQ, F(58), qPf, nullptr, BN, 256, 0);

  // ---- pointer scores + top-8 + q_attack ----
  pointer_final_kernel<<<BN / 8, 256, 0, stream>>>(qPf, kP_h, enemy_feats, maskEf,
                                                   ab, out_q);
}
